// MPNNPOM_36644660970032
// MI455X (gfx1250) — matrix operand order repaired and checked by reading, not compiled, not run on hardware
//
#include <hip/hip_runtime.h>

typedef __attribute__((ext_vector_type(16))) _Float16 v16h;
typedef __attribute__((ext_vector_type(8)))  float    v8f;
typedef __attribute__((ext_vector_type(2)))  float    f32x2;
typedef __attribute__((ext_vector_type(2)))  _Float16 f16x2;

#define NNODES 32768
#define NEDGES 65536
#define NGR    1024
#define DNODE  134
#define DEDGE  6
#define DH     100
#define DEH    128
#define DMSG   200     // 2*DH
#define NTASK  138
#define BK     64      // K-tile (2 WMMA k-steps per stage)

__device__ __forceinline__ float sigf(float x) { return 1.f / (1.f + expf(-x)); }

// ---------------------------------------------------------------------------
// WMMA GEMM:  C[M,N] = act( A[M,K] @ B[K,N] + bias + (beta ? C : 0) )
// fp32 in/out, f16 tiles in LDS, f32 accumulate (v_wmma_f32_16x16x32_f16).
// Block = 4 waves (32x4); block tile = 64 rows x 32 cols; each wave owns a
// 16x32 slice (two 16x16 accumulators).  K staged in 64-chunks -> 4 WMMA
// issues per barrier round.  Double-buffered LDS: one barrier per k-tile.
// Fast path batches all global loads into registers before LDS stores so
// the memory system sees 24 outstanding loads instead of 1.
// ---------------------------------------------------------------------------
__global__ __launch_bounds__(128)
void gemm_wmma(const float* __restrict__ A, const float* __restrict__ B,
               const float* __restrict__ bias, float* __restrict__ C,
               int M, int K, int N, int relu, int beta)
{
  __shared__ _Float16 As[2][4][16][BK];   // ping-pong [buf][wave][row][k] 16 KB
  __shared__ _Float16 Bs[2][32][BK];      // ping-pong [buf][col][k]        8 KB
  const int lane = threadIdx.x;
  const int wave = threadIdx.y;
  const int tid  = wave * 32 + lane;
  const int n0   = blockIdx.x * 32;
  const int mblk = blockIdx.y * 64;
  const int m0   = mblk + wave * 16;

  v8f acc0 = {0.f,0.f,0.f,0.f,0.f,0.f,0.f,0.f};
  v8f acc1 = {0.f,0.f,0.f,0.f,0.f,0.f,0.f,0.f};

  // uniform across the block: all K/N in this model are even -> f32x2 aligned
  const bool fast_mn = ((K & 1) == 0) && ((N & 1) == 0) &&
                       (mblk + 64 <= M) && (n0 + 32 <= N);

  int p = 0;
  for (int k0 = 0; k0 < K; k0 += BK, p ^= 1) {
    if (fast_mn && (k0 + BK <= K)) {
      // ---- fast path: batch loads into registers, then convert + store ----
      f32x2 av[16];                        // A tile 64x64: 16 f32x2 per thread
      #pragma unroll
      for (int i = 0; i < 16; ++i) {
        int slot = tid + i * 128;
        int r  = slot >> 5;                // 32 f32x2 per row
        int kk = (slot & 31) * 2;
        av[i] = *(const f32x2*)(A + (size_t)(mblk + r) * K + (k0 + kk));
      }
      f32x2 bv[8];                         // B tile 64x32: 8 f32x2 per thread
      #pragma unroll
      for (int i = 0; i < 8; ++i) {
        int slot = tid + i * 128;
        int kk   = slot >> 4;              // 16 f32x2 per k-row
        int nseg = (slot & 15) * 2;
        bv[i] = *(const f32x2*)(B + (size_t)(k0 + kk) * N + (n0 + nseg));
      }
      if (k0 + BK < K) {
        int gm = mblk + (tid >> 1);
        __builtin_prefetch(&A[(size_t)gm * K + k0 + BK], 0, 1);  // global_prefetch_b8
      }
      #pragma unroll
      for (int i = 0; i < 16; ++i) {
        int slot = tid + i * 128;
        int r  = slot >> 5;
        int kk = (slot & 31) * 2;
        f16x2 hv; hv.x = (_Float16)av[i].x; hv.y = (_Float16)av[i].y;
        *(f16x2*)&As[p][r >> 4][r & 15][kk] = hv;
      }
      #pragma unroll
      for (int i = 0; i < 8; ++i) {
        int slot = tid + i * 128;
        int kk   = slot >> 4;
        int nseg = (slot & 15) * 2;
        Bs[p][nseg][kk]     = (_Float16)bv[i].x;
        Bs[p][nseg + 1][kk] = (_Float16)bv[i].y;
      }
    } else {
      // ---- edge path: zero-fill LDS, then predicated scalar loads ----
      unsigned long long* Az = (unsigned long long*)&As[p][0][0][0];
      for (int i = tid; i < (4 * 16 * BK) / 4; i += 128) Az[i] = 0ull;
      unsigned long long* Bz = (unsigned long long*)&Bs[p][0][0];
      for (int i = tid; i < (32 * BK) / 4; i += 128) Bz[i] = 0ull;
      __syncthreads();
      for (int i = tid; i < 64 * BK; i += 128) {
        int r = i >> 6, kk = i & (BK - 1);
        int gm = mblk + r, gk = k0 + kk;
        if (gm < M && gk < K)
          As[p][r >> 4][r & 15][kk] = (_Float16)A[(size_t)gm * K + gk];
      }
      for (int i = tid; i < 32 * BK; i += 128) {
        int kk = i >> 5, c = i & 31;
        int gk = k0 + kk, gn = n0 + c;
        if (gk < K && gn < N)
          Bs[p][c][kk] = (_Float16)B[(size_t)gk * N + gn];
      }
    }
    __syncthreads();   // single barrier per k-tile (double buffering)

    // Fragments in contiguous k order (k = 16*(lane/16) + s) for A and B; two k-steps.
    const int arow  = lane & 15;
    const int abase = (lane < 16) ? 0 : 16;
    const int bcol  = lane & 15;
    const int bbase = (lane < 16) ? 0 : 16;
    #pragma unroll
    for (int kc = 0; kc < 2; ++kc) {
      union { v16h v; _Float16 h[16]; } af, bf0, bf1;
      #pragma unroll
      for (int hh = 0; hh < 16; ++hh)
        af.h[hh] = As[p][wave][arow][kc * 32 + hh + abase];
      #pragma unroll
      for (int hh = 0; hh < 16; ++hh) {
        bf0.h[hh] = Bs[p][bcol][kc * 32 + bbase + hh];
        bf1.h[hh] = Bs[p][bcol + 16][kc * 32 + bbase + hh];
      }
      acc0 = __builtin_amdgcn_wmma_f32_16x16x32_f16(false, af.v, false, bf0.v,
                                                    (short)0, acc0, false, false);
      acc1 = __builtin_amdgcn_wmma_f32_16x16x32_f16(false, af.v, false, bf1.v,
                                                    (short)0, acc1, false, false);
    }
  }

  // Epilogue: VGPR r -> M = r (+8 for lanes 16-31); N = lane%16 (+0/+16)
  union { v8f v; float f[8]; } c0, c1; c0.v = acc0; c1.v = acc1;
  const int rbase = m0 + ((lane < 16) ? 0 : 8);
  #pragma unroll
  for (int half = 0; half < 2; ++half) {
    const int col = n0 + (lane & 15) + half * 16;
    if (col >= N) continue;
    const float bv = bias ? bias[col] : 0.f;
    const float* cf = half ? c1.f : c0.f;
    #pragma unroll
    for (int r = 0; r < 8; ++r) {
      int row = rbase + r;
      if (row < M) {
        size_t idx = (size_t)row * N + col;
        float v = cf[r] + bv;
        if (beta) v += C[idx];
        if (relu && v < 0.f) v = 0.f;
        C[idx] = v;
      }
    }
  }
}

// ---------------------------------------------------------------------------
// Elementwise / gather / scatter helpers (grid-stride)
// ---------------------------------------------------------------------------
#define GS_LOOP(i, n) for (size_t i = blockIdx.x * (size_t)blockDim.x + threadIdx.x; \
                           i < (n); i += (size_t)gridDim.x * blockDim.x)

__global__ void fill_f32(float* p, float v, size_t n) { GS_LOOP(i, n) p[i] = v; }

__global__ void gather_concat(const float* __restrict__ h, const float* __restrict__ eh,
                              const int* __restrict__ src, float* __restrict__ X,
                              int E, int dh, int de)
{
  const int D = dh + de;
  GS_LOOP(idx, (size_t)E * D) {
    int e = (int)(idx / D), d = (int)(idx % D);
    X[idx] = (d < dh) ? h[(size_t)src[e] * dh + d] : eh[(size_t)e * de + (d - dh)];
  }
}

__global__ void scatter_add(const float* __restrict__ msg, const int* __restrict__ dst,
                            float* __restrict__ out, int E, int D)
{
  GS_LOOP(idx, (size_t)E * D) {
    int e = (int)(idx / D), d = (int)(idx % D);
    atomicAdd(&out[(size_t)dst[e] * D + d], msg[idx]);
  }
}

__global__ void gru_update(const float* __restrict__ gx, const float* __restrict__ gh,
                           float* __restrict__ h, int nn)
{
  GS_LOOP(idx, (size_t)nn * DH) {
    int n = (int)(idx / DH), d = (int)(idx % DH);
    const float* gxr = gx + (size_t)n * 3 * DH;
    const float* ghr = gh + (size_t)n * 3 * DH;
    float r  = sigf(gxr[d] + ghr[d]);
    float z  = sigf(gxr[DH + d] + ghr[DH + d]);
    float cn = tanhf(gxr[2 * DH + d] + r * ghr[2 * DH + d]);
    float ho = h[idx];
    h[idx] = (1.f - z) * cn + z * ho + ho;
  }
}

__global__ void readout_scatter(const float* __restrict__ hemb, const float* __restrict__ eemb,
                                const int* __restrict__ src, const int* __restrict__ dst,
                                float* __restrict__ out, int E)
{
  GS_LOOP(idx, (size_t)E * DMSG) {
    int e = (int)(idx / DMSG), d = (int)(idx % DMSG);
    float v = (d < DH) ? hemb[(size_t)src[e] * DH + d] : eemb[(size_t)e * DH + (d - DH)];
    atomicAdd(&out[(size_t)dst[e] * DMSG + d], v);
  }
}

__global__ void lstm_gate(const float* __restrict__ g, float* __restrict__ c,
                          float* __restrict__ h, int B)
{
  GS_LOOP(idx, (size_t)B * DMSG) {
    int b = (int)(idx / DMSG), d = (int)(idx % DMSG);
    const float* gb = g + (size_t)b * 4 * DMSG;
    float gi = gb[d], gf = gb[DMSG + d], gg = gb[2 * DMSG + d], go = gb[3 * DMSG + d];
    float cn = sigf(gf) * c[idx] + sigf(gi) * tanhf(gg);
    c[idx] = cn;
    h[idx] = sigf(go) * tanhf(cn);
  }
}

__global__ void s2s_logits(const float* __restrict__ x, const float* __restrict__ q,
                           const int* __restrict__ batch, float* __restrict__ e, int nn)
{
  GS_LOOP(n, (size_t)nn) {
    const float* xr = x + n * DMSG;
    const float* qr = q + (size_t)batch[n] * DMSG;
    float s = 0.f;
    for (int d = 0; d < DMSG; ++d) s += xr[d] * qr[d];
    e[n] = s;
  }
}

__global__ void seg_maxf(const float* __restrict__ e, const int* __restrict__ batch,
                         float* __restrict__ m, int nn)
{
  GS_LOOP(n, (size_t)nn) {
    float v = e[n]; int g = batch[n];
    if (v >= 0.f) atomicMax((int*)&m[g], __float_as_int(v));
    else          atomicMin((unsigned int*)&m[g], __float_as_uint(v));
  }
}

__global__ void s2s_exp(const float* __restrict__ e, const float* __restrict__ m,
                        const int* __restrict__ batch, float* __restrict__ ex,
                        float* __restrict__ denom, int nn)
{
  GS_LOOP(n, (size_t)nn) {
    int g = batch[n];
    float v = expf(e[n] - m[g]);
    ex[n] = v;
    atomicAdd(&denom[g], v);
  }
}

__global__ void s2s_weighted(const float* __restrict__ x, const float* __restrict__ ex,
                             const float* __restrict__ denom, const int* __restrict__ batch,
                             float* __restrict__ r, int nn)
{
  GS_LOOP(idx, (size_t)nn * DMSG) {
    int n = (int)(idx / DMSG), d = (int)(idx % DMSG);
    int g = batch[n];
    atomicAdd(&r[(size_t)g * DMSG + d], (ex[n] / denom[g]) * x[idx]);
  }
}

__global__ void s2s_pack(const float* __restrict__ q, const float* __restrict__ r,
                         float* __restrict__ qs, int B)
{
  GS_LOOP(idx, (size_t)B * 2 * DMSG) {
    int g = (int)(idx / (2 * DMSG)), d = (int)(idx % (2 * DMSG));
    qs[idx] = (d < DMSG) ? q[(size_t)g * DMSG + d] : r[(size_t)g * DMSG + (d - DMSG)];
  }
}

__global__ void enc_store(const float* __restrict__ qs, float* __restrict__ encs, int bb, int B)
{
  GS_LOOP(idx, (size_t)B * 400) {
    int g = (int)(idx / 400), d = (int)(idx % 400);
    encs[(size_t)g * 1200 + bb * 400 + d] = qs[idx];
  }
}

__global__ void attn_scores(const float* __restrict__ q, const float* __restrict__ k,
                            float* __restrict__ colsum, int B)
{
  GS_LOOP(g, (size_t)B) {
    float s[3][3];
    for (int i = 0; i < 3; ++i)
      for (int j = 0; j < 3; ++j) {
        const float* qi = q + ((size_t)g * 3 + i) * 400;
        const float* kj = k + ((size_t)g * 3 + j) * 400;
        float acc = 0.f;
        for (int t = 0; t < 400; ++t) acc += qi[t] * kj[t];
        s[i][j] = acc;
      }
    float cs[3] = {0.f, 0.f, 0.f};
    for (int i = 0; i < 3; ++i) {
      float mx = fmaxf(s[i][0], fmaxf(s[i][1], s[i][2]));
      float e0 = expf(s[i][0] - mx), e1 = expf(s[i][1] - mx), e2 = expf(s[i][2] - mx);
      float den = e0 + e1 + e2;
      cs[0] += e0 / den; cs[1] += e1 / den; cs[2] += e2 / den;
    }
    colsum[g * 3 + 0] = cs[0]; colsum[g * 3 + 1] = cs[1]; colsum[g * 3 + 2] = cs[2];
  }
}

__global__ void attn_combine(const float* __restrict__ colsum, const float* __restrict__ v,
                             float* __restrict__ out, int B)
{
  GS_LOOP(idx, (size_t)B * 400) {
    int g = (int)(idx / 400), d = (int)(idx % 400);
    float acc = 0.f;
    for (int j = 0; j < 3; ++j)
      acc += colsum[g * 3 + j] * v[((size_t)g * 3 + j) * 400 + d];
    out[idx] = acc;
  }
}

__global__ void relu_copy(const float* __restrict__ a, float* __restrict__ b, size_t n)
{ GS_LOOP(i, n) { float v = a[i]; b[i] = v > 0.f ? v : 0.f; } }

__global__ void sigmoid_copy(const float* __restrict__ a, float* __restrict__ b, size_t n)
{ GS_LOOP(i, n) b[i] = sigf(a[i]); }

// ---------------------------------------------------------------------------
// Host orchestration
// ---------------------------------------------------------------------------
static inline unsigned nb(size_t n) {
  size_t b = (n + 255) / 256;
  return (unsigned)(b > (1u << 20) ? (1u << 20) : b);
}
static void gemm(const float* A, const float* B, const float* bias, float* C,
                 int M, int K, int N, int relu, int beta, hipStream_t s)
{
  dim3 grid((unsigned)((N + 31) / 32), (unsigned)((M + 63) / 64));
  dim3 blk(32, 4);
  gemm_wmma<<<grid, blk, 0, s>>>(A, B, bias, C, M, K, N, relu, beta);
}

extern "C" void kernel_launch(void* const* d_in, const int* in_sizes, int n_in,
                              void* d_out, int out_size, void* d_ws, size_t ws_size,
                              hipStream_t stream)
{
  (void)in_sizes; (void)n_in; (void)out_size; (void)ws_size;
  // JAX pytree flatten order: dict keys sorted, lists in order.
  // 0..4 raw inputs; 5..10 attn{Wk,Wq,Wv,bk,bq,bv}; 11..40 backbones[0..2]
  // {We,Wm,Wp,be,bm,bp,gWh,gWx,gbh,gbx}; 41..46 ffn{W0,W1,W2,b0,b1,b2};
  // 47..48 readout{W,b}; 49..57 s2s{Wh[0..2],Wx[0..2],b[0..2]}.
  auto in = [&](int i) { return (const float*)d_in[i]; };
  const float* node_feats = in(0);
  const float* edge_feats = in(1);
  const int* esrc = (const int*)d_in[2];
  const int* edst = (const int*)d_in[3];
  const int* ngrp = (const int*)d_in[4];
  const float *aWk = in(5), *aWq = in(6), *aWv = in(7), *abk = in(8), *abq = in(9), *abv = in(10);
  const float *fW0 = in(41), *fW1 = in(42), *fW2 = in(43), *fb0 = in(44), *fb1 = in(45), *fb2 = in(46);
  const float *rW = in(47), *rb = in(48);

  char* base = (char*)d_ws;
  size_t off = 0;
  auto alloc = [&](size_t nfl) -> float* {
    float* p = (float*)(base + off);
    off += ((nfl * 4 + 255) & ~(size_t)255);
    return p;
  };
  float* encs  = alloc((size_t)NGR * 1200);        // persists across backbones
  float* h     = alloc((size_t)NNODES * DH);
  float* eh    = alloc((size_t)NEDGES * DEH);
  size_t x_off = off;
  float* X     = alloc((size_t)NEDGES * (DH + DEH));
  float* msg   = alloc((size_t)NEDGES * DH);
  float* agg   = alloc((size_t)NNODES * DH);
  float* gx    = alloc((size_t)NNODES * 3 * DH);
  float* gh    = alloc((size_t)NNODES * 3 * DH);
  float* nsum  = alloc((size_t)NNODES * DMSG);
  float* eemb  = alloc((size_t)NEDGES * DH);
  float* qstar = alloc((size_t)NGR * 2 * DMSG);
  float* hs    = alloc((size_t)3 * NGR * DMSG);
  float* cs    = alloc((size_t)3 * NGR * DMSG);
  float* gbuf  = alloc((size_t)NGR * 4 * DMSG);
  float* elog  = alloc(NNODES);
  float* exb   = alloc(NNODES);
  float* mbuf  = alloc(NGR);
  float* denom = alloc(NGR);
  float* rbuf  = alloc((size_t)NGR * DMSG);

  const float NEG_INF = -__builtin_huge_valf();

  for (int b = 0; b < 3; ++b) {
    const int pb = 11 + 10 * b;
    const float *We = in(pb), *Wm = in(pb + 1), *Wp = in(pb + 2);
    const float *be = in(pb + 3), *bm = in(pb + 4), *bp = in(pb + 5);
    const float *gWh = in(pb + 6), *gWx = in(pb + 7), *gbh = in(pb + 8), *gbx = in(pb + 9);

    gemm(node_feats, Wp, bp, h, NNODES, DNODE, DH, 1, 0, stream);
    gemm(edge_feats, We, be, eh, NEDGES, DEDGE, DEH, 1, 0, stream);

    for (int step = 0; step < 3; ++step) {
      gather_concat<<<nb((size_t)NEDGES * (DH + DEH)), 256, 0, stream>>>(h, eh, esrc, X, NEDGES, DH, DEH);
      gemm(X, Wm, bm, msg, NEDGES, DH + DEH, DH, 1, 0, stream);
      fill_f32<<<nb((size_t)NNODES * DH), 256, 0, stream>>>(agg, 0.f, (size_t)NNODES * DH);
      scatter_add<<<nb((size_t)NEDGES * DH), 256, 0, stream>>>(msg, edst, agg, NEDGES, DH);
      gemm(agg, gWx, gbx, gx, NNODES, DH, 3 * DH, 0, 0, stream);
      gemm(h,   gWh, gbh, gh, NNODES, DH, 3 * DH, 0, 0, stream);
      gru_update<<<nb((size_t)NNODES * DH), 256, 0, stream>>>(gx, gh, h, NNODES);
    }

    gemm(edge_feats, rW, rb, eemb, NEDGES, DEDGE, DH, 1, 0, stream);
    fill_f32<<<nb((size_t)NNODES * DMSG), 256, 0, stream>>>(nsum, 0.f, (size_t)NNODES * DMSG);
    readout_scatter<<<nb((size_t)NEDGES * DMSG), 256, 0, stream>>>(h, eemb, esrc, edst, nsum, NEDGES);

    fill_f32<<<nb((size_t)NGR * 2 * DMSG), 256, 0, stream>>>(qstar, 0.f, (size_t)NGR * 2 * DMSG);
    fill_f32<<<nb((size_t)3 * NGR * DMSG), 256, 0, stream>>>(hs, 0.f, (size_t)3 * NGR * DMSG);
    fill_f32<<<nb((size_t)3 * NGR * DMSG), 256, 0, stream>>>(cs, 0.f, (size_t)3 * NGR * DMSG);
    for (int it = 0; it < 6; ++it) {
      const float* inp = qstar; int din = 2 * DMSG;
      for (int l = 0; l < 3; ++l) {
        const float* Wx = in(52 + l);
        const float* Wh = in(49 + l);
        const float* bl = in(55 + l);
        float* hl = hs + (size_t)l * NGR * DMSG;
        float* cl = cs + (size_t)l * NGR * DMSG;
        gemm(inp, Wx, bl, gbuf, NGR, din, 4 * DMSG, 0, 0, stream);
        gemm(hl,  Wh, nullptr, gbuf, NGR, DMSG, 4 * DMSG, 0, 1, stream);
        lstm_gate<<<nb((size_t)NGR * DMSG), 256, 0, stream>>>(gbuf, cl, hl, NGR);
        inp = hl; din = DMSG;
      }
      const float* q = hs + (size_t)2 * NGR * DMSG;
      s2s_logits<<<nb(NNODES), 256, 0, stream>>>(nsum, q, ngrp, elog, NNODES);
      fill_f32<<<nb(NGR), 256, 0, stream>>>(mbuf, NEG_INF, NGR);
      seg_maxf<<<nb(NNODES), 256, 0, stream>>>(elog, ngrp, mbuf, NNODES);
      fill_f32<<<nb(NGR), 256, 0, stream>>>(denom, 0.f, NGR);
      s2s_exp<<<nb(NNODES), 256, 0, stream>>>(elog, mbuf, ngrp, exb, denom, NNODES);
      fill_f32<<<nb((size_t)NGR * DMSG), 256, 0, stream>>>(rbuf, 0.f, (size_t)NGR * DMSG);
      s2s_weighted<<<nb((size_t)NNODES * DMSG), 256, 0, stream>>>(nsum, exb, denom, ngrp, rbuf, NNODES);
      s2s_pack<<<nb((size_t)NGR * 2 * DMSG), 256, 0, stream>>>(q, rbuf, qstar, NGR);
    }
    enc_store<<<nb((size_t)NGR * 400), 256, 0, stream>>>(qstar, encs, b, NGR);
  }

  // attention stage reuses dead backbone scratch
  off = x_off;
  float* qb       = alloc((size_t)3 * NGR * 400);
  float* kb       = alloc((size_t)3 * NGR * 400);
  float* vb       = alloc((size_t)3 * NGR * 400);
  float* colsum   = alloc((size_t)NGR * 3);
  float* combined = alloc((size_t)NGR * 400);
  float* f0       = alloc((size_t)NGR * 300);
  float* embrelu  = alloc((size_t)NGR * 256);

  gemm(encs, aWq, abq, qb, 3 * NGR, 400, 400, 0, 0, stream);
  gemm(encs, aWk, abk, kb, 3 * NGR, 400, 400, 0, 0, stream);
  gemm(encs, aWv, abv, vb, 3 * NGR, 400, 400, 0, 0, stream);
  attn_scores<<<nb(NGR), 256, 0, stream>>>(qb, kb, colsum, NGR);
  attn_combine<<<nb((size_t)NGR * 400), 256, 0, stream>>>(colsum, vb, combined, NGR);

  float* out_proba  = (float*)d_out;
  float* out_logits = out_proba + (size_t)NGR * NTASK;
  float* out_emb    = out_logits + (size_t)NGR * NTASK;
  gemm(combined, fW0, fb0, f0, NGR, 400, 300, 1, 0, stream);
  gemm(f0, fW1, fb1, out_emb, NGR, 300, 256, 0, 0, stream);
  relu_copy<<<nb((size_t)NGR * 256), 256, 0, stream>>>(out_emb, embrelu, (size_t)NGR * 256);
  gemm(embrelu, fW2, fb2, out_logits, NGR, 256, NTASK, 0, 0, stream);
  sigmoid_copy<<<nb((size_t)NGR * NTASK), 256, 0, stream>>>(out_logits, out_proba, (size_t)NGR * NTASK);
}
